// GRU_54932631716305
// MI455X (gfx1250) — matrix operand order repaired and checked by reading, not compiled, not run on hardware
//
#include <hip/hip_runtime.h>
#include <hip/hip_bf16.h>

// Problem sizes (fixed by the reference)
constexpr int NB = 64;     // batch
constexpr int NT = 512;    // time steps
constexpr int NI = 256;    // input features
constexpr int NH = 512;    // hidden
constexpr int NG = 3 * NH; // 1536 gate dim

typedef __attribute__((ext_vector_type(16))) __bf16 v16bf;
typedef __attribute__((ext_vector_type(8)))  __bf16 v8bf;
typedef __attribute__((ext_vector_type(8)))  float  v8f;

union ABf { v16bf v; v8bf h[2]; };

// D(f32 16x16) = A(bf16 16x32) * B(bf16 32x16) + C
#define WMMA_BF16(a, b, c) \
  __builtin_amdgcn_wmma_f32_16x16x32_bf16(false, (a), false, (b), (short)0, (c), false, false)

// ---------------------------------------------------------------------------
// f32 -> bf16 conversion
// ---------------------------------------------------------------------------
__global__ void cvt_bf16_kernel(const float* __restrict__ in,
                                __bf16* __restrict__ out, int n) {
  int i = blockIdx.x * blockDim.x + threadIdx.x;
  if (i < n) out[i] = (__bf16)in[i];
}

// zero h0 (f32 + bf16 views)
__global__ void zero_h_kernel(float* __restrict__ hf, __bf16* __restrict__ hb) {
  int i = blockIdx.x * blockDim.x + threadIdx.x;
  if (i < NB * NH) { hf[i] = 0.0f; hb[i] = (__bf16)0.0f; }
}

// ---------------------------------------------------------------------------
// x_proj = x @ W_ih^T + b_ih  : (B*T, I=256) x (I, 3H) -> (B*T, 3H) f32
// One wave computes a 16(M) x 64(N) tile; K=256 (8 wmma k-steps, A reused x4).
// ---------------------------------------------------------------------------
__global__ void xproj_kernel(const __bf16* __restrict__ xb,
                             const __bf16* __restrict__ wihb,
                             const float* __restrict__ bih,
                             float* __restrict__ xproj) {
  const int wave = (blockIdx.x * blockDim.x + threadIdx.x) >> 5;
  const int lane = threadIdx.x & 31;
  const int lmod = lane & 15, lhi = lane >> 4;
  const int mt  = wave / 24;       // 0..2047 (M tiles of 16 over B*T)
  const int nt4 = wave % 24;       // 0..23   (N tiles of 64 over 3H)
  const int m0 = mt * 16, n0 = nt4 * 64;

  v8f acc[4] = {v8f{}, v8f{}, v8f{}, v8f{}};
  const __bf16* abase = xb + (size_t)(m0 + lmod) * NI;

  for (int kt = 0; kt < NI / 32; ++kt) {
    const int k0 = kt * 32;
    ABf a;
    const v8bf* pa = (const v8bf*)(abase + k0 + lhi * 16);
    a.h[0] = pa[0];        // K = k0 + lhi*16 + (0..7)
    a.h[1] = pa[1];        // K = k0 + lhi*16 + 8 + (0..7)
#pragma unroll
    for (int j = 0; j < 4; ++j) {
      const int n = n0 + j * 16 + lmod;
      const v16bf* pb = (const v16bf*)(wihb + (size_t)n * NI + k0 + lhi * 16);
      acc[j] = WMMA_BF16(a.v, *pb, acc[j]);
    }
  }
#pragma unroll
  for (int j = 0; j < 4; ++j) {
    const int n = n0 + j * 16 + lmod;
    const float bias = bih[n];
#pragma unroll
    for (int v = 0; v < 8; ++v) {
      const int m = m0 + v + lhi * 8;
      xproj[(size_t)m * NG + n] = acc[j][v] + bias;
    }
  }
}

// ---------------------------------------------------------------------------
// One GRU timestep. Grid: 128 blocks (4 row-tiles x 32 col-tiles of H),
// 128 threads = 4 waves. Each wave covers K-slice of 128 (4 wmma k-steps)
// for the coupled r/z/n accumulators; LDS reduce; wave 0 fuses the gates.
// h ping-pong: reads hin (f32 elementwise + bf16 matmul), writes hout + hs.
// ---------------------------------------------------------------------------
__global__ void gru_step_kernel(const __bf16* __restrict__ hin_b,
                                const float*  __restrict__ hin_f,
                                const __bf16* __restrict__ whhb,
                                const float*  __restrict__ bhh,
                                const float*  __restrict__ xproj, int t,
                                float*  __restrict__ hout_f,
                                __bf16* __restrict__ hout_b,
                                float*  __restrict__ hs_f,
                                __bf16* __restrict__ hs_b) {
  __shared__ float red[4][3][256];

  const int tile = blockIdx.x;          // 0..127
  const int rt = tile >> 5;             // batch row tile 0..3
  const int c  = tile & 31;             // H col tile 0..31
  const int w    = threadIdx.x >> 5;    // wave 0..3 -> K slice
  const int lane = threadIdx.x & 31;
  const int lmod = lane & 15, lhi = lane >> 4;

  v8f aR = {}, aZ = {}, aN = {};
  const __bf16* abase = hin_b + (size_t)(rt * 16 + lmod) * NH;
  const int nrow = c * 16 + lmod;
  const __bf16* brb = whhb + (size_t)nrow * NH;
  const __bf16* bzb = whhb + (size_t)(NH + nrow) * NH;
  const __bf16* bnb = whhb + (size_t)(2 * NH + nrow) * NH;

#pragma unroll
  for (int kk = 0; kk < 4; ++kk) {
    const int k0 = (w * 4 + kk) * 32;
    ABf a;
    const v8bf* pa = (const v8bf*)(abase + k0 + lhi * 16);
    a.h[0] = pa[0];
    a.h[1] = pa[1];
    aR = WMMA_BF16(a.v, *(const v16bf*)(brb + k0 + lhi * 16), aR);
    aZ = WMMA_BF16(a.v, *(const v16bf*)(bzb + k0 + lhi * 16), aZ);
    aN = WMMA_BF16(a.v, *(const v16bf*)(bnb + k0 + lhi * 16), aN);
  }
#pragma unroll
  for (int v = 0; v < 8; ++v) {
    red[w][0][v * 32 + lane] = aR[v];
    red[w][1][v * 32 + lane] = aZ[v];
    red[w][2][v * 32 + lane] = aN[v];
  }
  __syncthreads();

  if (w == 0) {
    const int n = c * 16 + lmod;        // column within H
    const float bR = bhh[n], bZ = bhh[NH + n], bN = bhh[2 * NH + n];
#pragma unroll
    for (int v = 0; v < 8; ++v) {
      const int p = v * 32 + lane;
      const float hr = red[0][0][p] + red[1][0][p] + red[2][0][p] + red[3][0][p] + bR;
      const float hz = red[0][1][p] + red[1][1][p] + red[2][1][p] + red[3][1][p] + bZ;
      const float hn = red[0][2][p] + red[1][2][p] + red[2][2][p] + red[3][2][p] + bN;
      const int m = rt * 16 + v + lhi * 8;       // batch index
      const float* xp = xproj + ((size_t)m * NT + t) * NG;
      const float xr = xp[n], xz = xp[NH + n], xn = xp[2 * NH + n];
      const float r  = 1.0f / (1.0f + __expf(-(xr + hr)));
      const float z  = 1.0f / (1.0f + __expf(-(xz + hz)));
      const float nn = tanhf(xn + r * hn);
      const float hp = hin_f[(size_t)m * NH + n];
      const float hnew = (1.0f - z) * nn + z * hp;
      hout_f[(size_t)m * NH + n] = hnew;
      hout_b[(size_t)m * NH + n] = (__bf16)hnew;
      const size_t hso = ((size_t)t * NB + m) * NH + n;
      hs_f[hso] = hnew;
      hs_b[hso] = (__bf16)hnew;
    }
  }
}

// ---------------------------------------------------------------------------
// scores[t*B+b] = sum_h attn_v[h] * tanh( (hs @ attn_W^T)[t,b,h] + attn_b[h] )
// One wave per 16 rows of (T*B); loops all 32 N-tiles, K=512; fused epilogue
// with half-wave (16 lane) shuffle reduction.
// ---------------------------------------------------------------------------
__global__ void attn_scores_kernel(const __bf16* __restrict__ hsb,
                                   const __bf16* __restrict__ awb,
                                   const float* __restrict__ ab,
                                   const float* __restrict__ av,
                                   float* __restrict__ scores) {
  const int wave = (blockIdx.x * blockDim.x + threadIdx.x) >> 5;  // 0..2047
  const int lane = threadIdx.x & 31;
  const int lmod = lane & 15, lhi = lane >> 4;
  const int m0 = wave * 16;
  const __bf16* abase = hsb + (size_t)(m0 + lmod) * NH;

  float part[8] = {0.f, 0.f, 0.f, 0.f, 0.f, 0.f, 0.f, 0.f};

  for (int nt = 0; nt < NH / 16; ++nt) {
    v8f acc = {};
    const int n = nt * 16 + lmod;
    const __bf16* bbase = awb + (size_t)n * NH;
    for (int kt = 0; kt < NH / 32; ++kt) {
      const int k0 = kt * 32;
      ABf a;
      const v8bf* pa = (const v8bf*)(abase + k0 + lhi * 16);
      a.h[0] = pa[0];
      a.h[1] = pa[1];
      acc = WMMA_BF16(a.v, *(const v16bf*)(bbase + k0 + lhi * 16), acc);
    }
    const float bb = ab[n], vv = av[n];
#pragma unroll
    for (int v = 0; v < 8; ++v) part[v] += tanhf(acc[v] + bb) * vv;
  }
#pragma unroll
  for (int v = 0; v < 8; ++v) {
    float s = part[v];
    for (int msk = 8; msk >= 1; msk >>= 1) s += __shfl_xor(s, msk, 16);
    if (lmod == 0) scores[m0 + lhi * 8 + v] = s;  // rows m0+v / m0+8+v
  }
}

// ---------------------------------------------------------------------------
// Per-batch softmax over T + weighted context sum. One block per b.
// ---------------------------------------------------------------------------
__global__ void softmax_context_kernel(const float* __restrict__ scores,
                                       const float* __restrict__ hs,
                                       float* __restrict__ out) {
  const int b = blockIdx.x;
  const int tid = threadIdx.x;  // 256 threads
  __shared__ float wgt[NT];
  __shared__ float red[256];

  const float s0 = scores[(size_t)tid * NB + b];
  const float s1 = scores[(size_t)(tid + 256) * NB + b];

  red[tid] = fmaxf(s0, s1);
  __syncthreads();
  for (int s = 128; s > 0; s >>= 1) {
    if (tid < s) red[tid] = fmaxf(red[tid], red[tid + s]);
    __syncthreads();
  }
  const float mx = red[0];
  __syncthreads();

  const float e0 = __expf(s0 - mx), e1 = __expf(s1 - mx);
  red[tid] = e0 + e1;
  __syncthreads();
  for (int s = 128; s > 0; s >>= 1) {
    if (tid < s) red[tid] += red[tid + s];
    __syncthreads();
  }
  const float inv = 1.0f / red[0];
  wgt[tid] = e0 * inv;
  wgt[tid + 256] = e1 * inv;
  __syncthreads();

  for (int h = tid; h < NH; h += 256) {
    float acc = 0.0f;
    for (int t = 0; t < NT; ++t)
      acc += wgt[t] * hs[((size_t)t * NB + b) * NH + h];
    out[(size_t)b * NH + h] = acc;
  }
}

// ---------------------------------------------------------------------------
// Launch
// ---------------------------------------------------------------------------
extern "C" void kernel_launch(void* const* d_in, const int* in_sizes, int n_in,
                              void* d_out, int out_size, void* d_ws, size_t ws_size,
                              hipStream_t stream) {
  (void)in_sizes; (void)n_in; (void)out_size; (void)ws_size;

  const float* x   = (const float*)d_in[0];  // (B,T,I)
  const float* wih = (const float*)d_in[1];  // (3H,I)
  const float* whh = (const float*)d_in[2];  // (3H,H)
  const float* bih = (const float*)d_in[3];  // (3H)
  const float* bhh = (const float*)d_in[4];  // (3H)
  const float* aW  = (const float*)d_in[5];  // (H,H)
  const float* ab  = (const float*)d_in[6];  // (H)
  const float* av  = (const float*)d_in[7];  // (H)
  float* out = (float*)d_out;                // (B,H)

  char* p = (char*)d_ws;
  auto alloc = [&](size_t bytes) -> char* {
    char* r = p;
    p += (bytes + 255) & ~(size_t)255;
    return r;
  };

  const int NXB = NB * NT * NI;       // x elements
  __bf16* xb     = (__bf16*)alloc((size_t)NXB * 2);
  __bf16* wihb   = (__bf16*)alloc((size_t)NG * NI * 2);
  __bf16* whhb   = (__bf16*)alloc((size_t)NG * NH * 2);
  __bf16* awb    = (__bf16*)alloc((size_t)NH * NH * 2);
  float*  xproj  = (float*) alloc((size_t)NB * NT * NG * 4);
  float*  hf0    = (float*) alloc((size_t)NB * NH * 4);
  float*  hf1    = (float*) alloc((size_t)NB * NH * 4);
  __bf16* hb0    = (__bf16*)alloc((size_t)NB * NH * 2);
  __bf16* hb1    = (__bf16*)alloc((size_t)NB * NH * 2);
  float*  hsf    = (float*) alloc((size_t)NT * NB * NH * 4);
  __bf16* hsb    = (__bf16*)alloc((size_t)NT * NB * NH * 2);
  float*  scores = (float*) alloc((size_t)NT * NB * 4);

  // bf16 conversions of all matmul operands
  cvt_bf16_kernel<<<(NXB + 255) / 256, 256, 0, stream>>>(x, xb, NXB);
  cvt_bf16_kernel<<<(NG * NI + 255) / 256, 256, 0, stream>>>(wih, wihb, NG * NI);
  cvt_bf16_kernel<<<(NG * NH + 255) / 256, 256, 0, stream>>>(whh, whhb, NG * NH);
  cvt_bf16_kernel<<<(NH * NH + 255) / 256, 256, 0, stream>>>(aW, awb, NH * NH);
  zero_h_kernel<<<(NB * NH + 255) / 256, 256, 0, stream>>>(hf0, hb0);

  // x_proj GEMM: (B*T/16) M-tiles * (3H/64) N-tile-groups waves, 8 waves/block
  {
    const int waves = (NB * NT / 16) * (NG / 64);  // 49152
    xproj_kernel<<<waves / 8, 256, 0, stream>>>(xb, wihb, bih, xproj);
  }

  // Sequential scan: 512 step kernels (graph replay amortizes launch cost)
  float*  hfp[2] = {hf0, hf1};
  __bf16* hbp[2] = {hb0, hb1};
  for (int t = 0; t < NT; ++t) {
    gru_step_kernel<<<128, 128, 0, stream>>>(
        hbp[t & 1], hfp[t & 1], whhb, bhh, xproj, t,
        hfp[(t + 1) & 1], hbp[(t + 1) & 1], hsf, hsb);
  }

  // Attention scores: 2048 waves, 8 waves/block
  attn_scores_kernel<<<256, 256, 0, stream>>>(hsb, awb, ab, av, scores);

  // Softmax over T + context
  softmax_context_kernel<<<NB, 256, 0, stream>>>(scores, hsf, out);
}
